// SpeakerCrossAttention_69002944578189
// MI455X (gfx1250) — matrix operand order repaired and checked by reading, not compiled, not run on hardware
//
#include <hip/hip_runtime.h>
#include <math.h>

// ---------------- problem constants ----------------
#define B_DIM 16
#define D_DIM 512
#define T_DIM 8192
#define S_DIM 256
#define EPS_LN 1e-5f

// ---------------- WMMA tile config -----------------
#define MT    64              // tokens per block
#define KC    32              // K per WMMA step (bf16)
#define NSTEP (D_DIM / KC)    // 16 K-steps
#define LDA   520             // padded LDS row stride (elems): 260 dwords == 4 mod 64 banks
#define NT_TOTAL (D_DIM / 16) // 32 n-tiles, 4 per wave across 8 waves

typedef __attribute__((ext_vector_type(16))) __bf16 v16bf;
typedef __attribute__((ext_vector_type(8)))  float  v8f;

union Frag16 {
    v16bf v;
    float4 f4[2];
};

// =====================================================================
// Kernel 1: per-batch speaker chain -> attn_n, attnbias  (tiny GEMVs)
// =====================================================================
__global__ __launch_bounds__(256) void spk_prep_kernel(
    const float* __restrict__ spk_emb, const float* __restrict__ Wsp,
    const float* __restrict__ bsp,     const float* __restrict__ Wv,
    const float* __restrict__ bv,      const float* __restrict__ Wo,
    const float* __restrict__ bo,      const float* __restrict__ gamma,
    const float* __restrict__ beta,    const float* __restrict__ Wg1,
    const float* __restrict__ bg1,
    float* __restrict__ attnN_g, float* __restrict__ attnB_g)
{
    __shared__ float red[256];
    __shared__ float spkn[S_DIM];
    __shared__ float sf[D_DIM];
    __shared__ float vv[D_DIM];
    __shared__ float at[D_DIM];
    __shared__ float an[D_DIM];

    const int tid = threadIdx.x;
    const int b   = blockIdx.x;

    // ---- normalize speaker embedding (S=256) ----
    float se = spk_emb[b * S_DIM + tid];
    red[tid] = se * se;
    __syncthreads();
    for (int s = 128; s > 0; s >>= 1) {
        if (tid < s) red[tid] += red[tid + s];
        __syncthreads();
    }
    float nrm = fmaxf(sqrtf(red[0]), 1e-12f);
    __syncthreads();
    spkn[tid] = se / nrm;
    __syncthreads();

    // ---- speaker_feat = spk @ Wsp.T + bsp ----
    for (int d = tid; d < D_DIM; d += 256) {
        float acc = bsp[d];
        const float* w = Wsp + (size_t)d * S_DIM;
        for (int s = 0; s < S_DIM; ++s) acc += spkn[s] * w[s];
        sf[d] = acc;
    }
    __syncthreads();

    // ---- v = sf @ Wv.T + bv ----
    for (int d = tid; d < D_DIM; d += 256) {
        float acc = bv[d];
        const float* w = Wv + (size_t)d * D_DIM;
        for (int k = 0; k < D_DIM; ++k) acc += sf[k] * w[k];
        vv[d] = acc;
    }
    __syncthreads();

    // ---- attn = v @ Wo.T + bo ----
    for (int d = tid; d < D_DIM; d += 256) {
        float acc = bo[d];
        const float* w = Wo + (size_t)d * D_DIM;
        for (int k = 0; k < D_DIM; ++k) acc += vv[k] * w[k];
        at[d] = acc;
    }
    __syncthreads();

    // ---- LayerNorm over D ----
    float ls = 0.f;
    for (int d = tid; d < D_DIM; d += 256) ls += at[d];
    red[tid] = ls;
    __syncthreads();
    for (int s = 128; s > 0; s >>= 1) {
        if (tid < s) red[tid] += red[tid + s];
        __syncthreads();
    }
    float mu = red[0] / (float)D_DIM;
    __syncthreads();
    float lv = 0.f;
    for (int d = tid; d < D_DIM; d += 256) { float df = at[d] - mu; lv += df * df; }
    red[tid] = lv;
    __syncthreads();
    for (int s = 128; s > 0; s >>= 1) {
        if (tid < s) red[tid] += red[tid + s];
        __syncthreads();
    }
    float rstd = rsqrtf(red[0] / (float)D_DIM + EPS_LN);
    __syncthreads();
    for (int d = tid; d < D_DIM; d += 256) {
        float v = (at[d] - mu) * rstd * gamma[d] + beta[d];
        an[d] = v;
        attnN_g[b * D_DIM + d] = v;
    }
    __syncthreads();

    // ---- attnbias = attn_n @ W1a.T + bg1   (W1a = Wg1[:, D:]) ----
    for (int d = tid; d < D_DIM; d += 256) {
        float acc = bg1[d];
        const float* w = Wg1 + (size_t)d * (2 * D_DIM) + D_DIM;
        for (int k = 0; k < D_DIM; ++k) acc += an[k] * w[k];
        attnB_g[b * D_DIM + d] = acc;
    }
}

// =====================================================================
// Kernel 2: swizzle W1f = Wg1[:, :D] into bf16 WMMA B-fragment order.
// idx = ((ks*32 + nt)*32 + lane)*16 + j
//   lanes 0-15  : K = ks*32 + j      , N = nt*16 + lane
//   lanes 16-31 : K = ks*32 + 16 + j , N = nt*16 + lane - 16
// =====================================================================
__global__ __launch_bounds__(256) void wswz_kernel(
    const float* __restrict__ Wg1, __bf16* __restrict__ Bswz)
{
    int idx  = blockIdx.x * 256 + threadIdx.x;   // < 512*512
    int j    = idx & 15;
    int lane = (idx >> 4) & 31;
    int nt   = (idx >> 9) & 31;
    int ks   = idx >> 14;
    int n = nt * 16 + (lane & 15);
    int k = ks * KC + (lane >> 4) * 16 + j;
    Bswz[idx] = (__bf16)Wg1[(size_t)n * (2 * D_DIM) + k];
}

// =====================================================================
// Kernel 3: fused gate. grid = (T/MT, B), 256 threads = 8 wave32.
// Full 64x512 bf16 A tile resident in LDS: single staging barrier,
// barrier-free WMMA K-loop, epilogue re-reads features from LDS.
// =====================================================================
__global__ __launch_bounds__(256) void fused_gate_kernel(
    const float* __restrict__ features, const __bf16* __restrict__ Bswz,
    const float* __restrict__ attnN_g,  const float* __restrict__ attnB_g,
    const float* __restrict__ Wg2,      const float* __restrict__ bg2,
    float* __restrict__ out)
{
    __shared__ __align__(16) __bf16 sA[MT * LDA];   // 64 x 512(+8) bf16 ~ 65 KB
    __shared__ float sPart[8 * MT];
    __shared__ float sAlpha[MT];

    const int tid  = threadIdx.x;
    const int lane = tid & 31;
    const int wave = tid >> 5;
    const int b    = blockIdx.y;
    const int t0   = blockIdx.x * MT;

    const float* fbase = features + (size_t)b * D_DIM * T_DIM + t0;

    const int tloc = tid & 63;   // token within tile
    const int kr   = tid >> 6;   // k-row group (0..3)

    // ---- stage the full 64x512 tile (f32 -> bf16), streaming (non-temporal) ----
    for (int ks = 0; ks < NSTEP; ++ks) {
        float pf[8];
#pragma unroll
        for (int j = 0; j < 8; ++j)
            pf[j] = __builtin_nontemporal_load(
                &fbase[(size_t)(ks * KC + kr * 8 + j) * T_DIM + tloc]);
#pragma unroll
        for (int j = 0; j < 8; ++j)
            sA[tloc * LDA + ks * KC + kr * 8 + j] = (__bf16)pf[j];
    }
    __syncthreads();

    const int lh  = lane & 15;
    const int sel = lane >> 4;
    const int ntb = wave * 4;    // this wave's first n-tile

    const v8f zacc = {0.f, 0.f, 0.f, 0.f, 0.f, 0.f, 0.f, 0.f};
    v8f acc[4][4];
#pragma unroll
    for (int mt = 0; mt < 4; ++mt)
#pragma unroll
        for (int i = 0; i < 4; ++i) acc[mt][i] = zacc;

    // ---- barrier-free WMMA K-loop: A read-only in LDS, B streamed from L2 ----
    for (int ks = 0; ks < NSTEP; ++ks) {
        Frag16 bfr[4];
#pragma unroll
        for (int i = 0; i < 4; ++i) {
            const __bf16* p = Bswz + (((size_t)ks * NT_TOTAL + (ntb + i)) * 32 + lane) * 16;
            bfr[i].f4[0] = ((const float4*)p)[0];
            bfr[i].f4[1] = ((const float4*)p)[1];
        }

        Frag16 afr[4];
#pragma unroll
        for (int mt = 0; mt < 4; ++mt) {
            const __bf16* p = &sA[(mt * 16 + lh) * LDA + ks * KC + sel * 16];
            afr[mt].f4[0] = ((const float4*)p)[0];
            afr[mt].f4[1] = ((const float4*)(p + 8))[0];
        }

#pragma unroll
        for (int mt = 0; mt < 4; ++mt)
#pragma unroll
            for (int i = 0; i < 4; ++i)
                acc[mt][i] = __builtin_amdgcn_wmma_f32_16x16x32_bf16(
                    false, afr[mt].v, false, bfr[i].v,
                    (short)0, acc[mt][i], false, false);
    }

    // ---- epilogue: h = relu(acc + attnbias); alpha = sigmoid(h . Wg2 + bg2) ----
    float abias[4], w2[4];
#pragma unroll
    for (int i = 0; i < 4; ++i) {
        int n = (ntb + i) * 16 + lh;
        abias[i] = attnB_g[b * D_DIM + n];
        w2[i]    = Wg2[n];
    }

#pragma unroll
    for (int mt = 0; mt < 4; ++mt) {
        float p[8];
#pragma unroll
        for (int r = 0; r < 8; ++r) {
            float s = 0.f;
#pragma unroll
            for (int i = 0; i < 4; ++i) {
                float h = acc[mt][i][r] + abias[i];
                s += fmaxf(h, 0.f) * w2[i];
            }
            // reduce over the 16 lanes of this half (n dimension)
#pragma unroll
            for (int m = 1; m < 16; m <<= 1) s += __shfl_xor(s, m, 32);
            p[r] = s;
        }
        if (lh == 0) {
            // C/D layout: VGPR r -> M=r (lanes 0-15) / M=r+8 (lanes 16-31)
#pragma unroll
            for (int r = 0; r < 8; ++r)
                sPart[wave * MT + mt * 16 + sel * 8 + r] = p[r];
        }
    }
    __syncthreads();

    if (tid < MT) {
        float s = bg2[0];
#pragma unroll
        for (int w = 0; w < 8; ++w) s += sPart[w * MT + tid];
        sAlpha[tid] = 1.f / (1.f + __expf(-s));
    }
    __syncthreads();

    // ---- gated combine; features come back out of LDS (no HBM re-read),
    //      output written in native (B, D, T) layout, coalesced in t, NT stores ----
    const float al  = sAlpha[tloc];
    const float oma = 1.f - al;
    float* obase = out + (size_t)b * D_DIM * T_DIM + t0;
    for (int d = kr; d < D_DIM; d += 4) {
        float f  = (float)sA[tloc * LDA + d];
        float an = attnN_g[b * D_DIM + d];
        __builtin_nontemporal_store(al * f + oma * an,
                                    &obase[(size_t)d * T_DIM + tloc]);
    }
}

// =====================================================================
extern "C" void kernel_launch(void* const* d_in, const int* in_sizes, int n_in,
                              void* d_out, int out_size, void* d_ws, size_t ws_size,
                              hipStream_t stream) {
    const float* features = (const float*)d_in[0];
    const float* spk_emb  = (const float*)d_in[1];
    const float* Wsp      = (const float*)d_in[2];
    const float* bsp      = (const float*)d_in[3];
    const float* Wv       = (const float*)d_in[4];
    const float* bv       = (const float*)d_in[5];
    const float* Wo       = (const float*)d_in[6];
    const float* bo       = (const float*)d_in[7];
    const float* gamma    = (const float*)d_in[8];
    const float* beta     = (const float*)d_in[9];
    const float* Wg1      = (const float*)d_in[10];
    const float* bg1      = (const float*)d_in[11];
    const float* Wg2      = (const float*)d_in[12];
    const float* bg2      = (const float*)d_in[13];
    float* out = (float*)d_out;

    // workspace layout: attnN (B*D f32) | attnB (B*D f32) | Bswz (D*D bf16)
    float*  attnN = (float*)d_ws;
    float*  attnB = attnN + B_DIM * D_DIM;
    __bf16* Bswz  = (__bf16*)((char*)d_ws + 2ull * B_DIM * D_DIM * sizeof(float));

    spk_prep_kernel<<<B_DIM, 256, 0, stream>>>(
        spk_emb, Wsp, bsp, Wv, bv, Wo, bo, gamma, beta, Wg1, bg1, attnN, attnB);

    wswz_kernel<<<(D_DIM * D_DIM) / 256, 256, 0, stream>>>(Wg1, Bswz);

    fused_gate_kernel<<<dim3(T_DIM / MT, B_DIM), 256, 0, stream>>>(
        features, Bswz, attnN, attnB, Wg2, bg2, out);
}
